// Net_86517821210769
// MI455X (gfx1250) — matrix operand order repaired and checked by reading, not compiled, not run on hardware
//
#include <hip/hip_runtime.h>
#include <math.h>

typedef __attribute__((ext_vector_type(16))) __bf16        v16bf;
typedef __attribute__((ext_vector_type(8)))  float         v8f;
typedef __attribute__((ext_vector_type(8)))  unsigned int  v8u;

#define T_IN   500
#define T_OUT  469
#define CIN    80
#define COUT1  40
#define KW     64
#define NBATCH 1024
#define C3     41
#define POS    (T_OUT * NBATCH)        /* 480256 positions (t,n) */
#define W1PK   (CIN * 3 * 2 * 32 * 8)  /* 122880 dwords per half (hi/lo) */
#define W3PK   (3 * 2 * 32 * 8)        /* 1536 dwords per half */

#if __has_builtin(__builtin_amdgcn_global_load_async_to_lds_b128) && \
    __has_builtin(__builtin_amdgcn_s_wait_asynccnt)
#define HAVE_ASYNC 1
typedef int v4i __attribute__((vector_size(16)));
typedef __attribute__((address_space(1))) v4i g_v4i;   // global int4
typedef __attribute__((address_space(3))) v4i l_v4i;   // LDS int4

static __device__ __forceinline__ void async_cp16(const unsigned int* src, unsigned int* ldsdst) {
  __builtin_amdgcn_global_load_async_to_lds_b128(
      (g_v4i*)(unsigned int*)src, (l_v4i*)ldsdst, 0, 0);
}
#else
#define HAVE_ASYNC 0
#endif

static __device__ __forceinline__ unsigned short f2bf(float f) {
  __bf16 h = (__bf16)f;                 // RNE
  unsigned short u; __builtin_memcpy(&u, &h, 2); return u;
}
static __device__ __forceinline__ float bfu2f(unsigned short u) {
  unsigned int w = ((unsigned int)u) << 16;
  float f; __builtin_memcpy(&f, &w, 4); return f;
}

// ---------------------------------------------------------------------------
// One-shot weight packing for conv1: bf16 hi/lo pairs in the exact WMMA
// B-fragment layout.  wpk[ci][ct][h][lane][vv] (dword = K-pair), lo at +W1PK.
// ---------------------------------------------------------------------------
__global__ void pack_w1(const float* __restrict__ w, unsigned int* __restrict__ wpk) {
  const int i = blockIdx.x * blockDim.x + threadIdx.x;
  if (i >= W1PK) return;
  const int ci   = i / 1536;
  int rem        = i % 1536;
  const int ct   = rem / 512;  rem %= 512;
  const int h    = rem / 256;  rem %= 256;
  const int lane = rem / 8;
  const int vv   = rem % 8;
  const int g    = lane >> 4;
  const int c    = ct * 16 + (lane & 15);
  const int k    = h * 32 + ((vv < 4) ? 0 : 16) + g * 8 + 2 * (vv & 3);  // B slot K = (vv<4?0:16) + g*8 + 2*(vv&3)
  float w0 = 0.0f, w1 = 0.0f;
  if (c < COUT1) {
    w0 = w[(c * CIN + ci) * KW + k];
    w1 = w[(c * CIN + ci) * KW + k + 1];
  }
  const unsigned short h0 = f2bf(w0), h1 = f2bf(w1);
  const unsigned short l0 = f2bf(w0 - bfu2f(h0)), l1 = f2bf(w1 - bfu2f(h1));
  wpk[i]        = (unsigned int)h0 | ((unsigned int)h1 << 16);
  wpk[W1PK + i] = (unsigned int)l0 | ((unsigned int)l1 << 16);
}

// One-shot packing for conv3's constant band matrix B[kk][c2] = w3[kk-c2+32].
__global__ void pack_w3(const float* __restrict__ w3, unsigned int* __restrict__ bpk) {
  const int i = blockIdx.x * blockDim.x + threadIdx.x;
  if (i >= W3PK) return;
  int rem        = i;
  const int ct   = rem / 512;  rem %= 512;
  const int h    = rem / 256;  rem %= 256;
  const int lane = rem / 8;
  const int vv   = rem % 8;
  const int g    = lane >> 4;
  const int c2   = ct * 16 + (lane & 15);
  const int kk   = h * 32 + ((vv < 4) ? 0 : 16) + g * 8 + 2 * (vv & 3);
  float b0 = 0.0f, b1 = 0.0f;
  if (c2 < C3) {
    const int wi0 = kk - c2 + 32, wi1 = wi0 + 1;
    if (kk < COUT1 && wi0 >= 0 && wi0 < KW)     b0 = w3[wi0];
    if (kk + 1 < COUT1 && wi1 >= 0 && wi1 < KW) b1 = w3[wi1];
  }
  const unsigned short h0 = f2bf(b0), h1 = f2bf(b1);
  const unsigned short l0 = f2bf(b0 - bfu2f(h0)), l1 = f2bf(b1 - bfu2f(h1));
  bpk[i]        = (unsigned int)h0 | ((unsigned int)h1 << 16);
  bpk[W3PK + i] = (unsigned int)l0 | ((unsigned int)l1 << 16);
}

// ---------------------------------------------------------------------------
// conv1 implicit GEMM, split-bf16 (hi/lo) 3-WMMA scheme.
// Block: 128 threads = 4 waves; wave = one 16-position tile, 3 channel-tile
// accumulators.  Weight slabs double-buffered in LDS via async global->LDS
// (ASYNCcnt) so the next ci's 12KB streams in under the current 18 WMMAs.
// ---------------------------------------------------------------------------
__global__ __launch_bounds__(128) void conv1_wmma(const float* __restrict__ x,
                                                  const unsigned int* __restrict__ wpk,
                                                  float* __restrict__ y1) {
  __shared__ __align__(32) unsigned int wbuf[2][2][1536];  // [buf][hi/lo][frag] 24KB
  __shared__ float xwin[CIN * 127];                        // 40.6KB input window

  const int tb  = blockIdx.x;
  const int n   = blockIdx.y;
  const int t0  = tb * 64;
  const int tid = threadIdx.x;

  // stage input window: xwin[ci*127 + j] = x[n][ci][t0-16+j], zero-padded
  {
    const int tt = t0 - 16 + tid;
    const bool ok = (tid < 127) && (tt >= 0) && (tt < T_IN);
    for (int ci = 0; ci < CIN; ++ci)
      if (tid < 127)
        xwin[ci * 127 + tid] = ok ? x[(n * CIN + ci) * T_IN + tt] : 0.0f;
  }

  // stage packed weight slab for one ci into wbuf[buf]
  auto stage = [&](int ci, int buf) {
    const unsigned int* ghi = wpk + (size_t)ci * 1536;
    const unsigned int* glo = ghi + W1PK;
#if HAVE_ASYNC
    for (int j = tid; j < 384; j += 128) {                  // 384 x 16B per half
      async_cp16(ghi + j * 4, &wbuf[buf][0][j * 4]);
      async_cp16(glo + j * 4, &wbuf[buf][1][j * 4]);
    }
#else
    for (int j = tid; j < 1536; j += 128) {
      wbuf[buf][0][j] = ghi[j];
      wbuf[buf][1][j] = glo[j];
    }
#endif
  };

  const int pt   = tid >> 5;          // wave = position tile 0..3
  const int lane = tid & 31;
  const int g    = lane >> 4;
  const int l15  = lane & 15;

  v8f accs[3] = {{}, {}, {}};

  stage(0, 0);
  for (int ci = 0; ci < CIN; ++ci) {
    const int cur = ci & 1;
#if HAVE_ASYNC
    __builtin_amdgcn_s_wait_asynccnt(0);   // my async writes into wbuf[cur] done
#endif
    __syncthreads();                       // visible to all; wbuf[cur^1] free
    if (ci + 1 < CIN) stage(ci + 1, cur ^ 1);

#pragma unroll
    for (int h = 0; h < 2; ++h) {
      const int kb = h * 32;
      const float* xr = &xwin[ci * 127 + pt * 16 + l15 + kb];
      v16bf a_hi = {}, a_lo = {};
#pragma unroll
      for (int vv = 0; vv < 8; ++vv) {
        // A slot (g,vv) -> K = (vv<4?0:16) + g*8 + 2*(vv&3)
        const int ka = ((vv < 4) ? 0 : 16) + g * 8 + 2 * (vv & 3);
        const float a0 = xr[ka], a1 = xr[ka + 1];
        const __bf16 ah0 = (__bf16)a0, ah1 = (__bf16)a1;
        a_hi[2 * vv]     = ah0;
        a_hi[2 * vv + 1] = ah1;
        a_lo[2 * vv]     = (__bf16)(a0 - (float)ah0);
        a_lo[2 * vv + 1] = (__bf16)(a1 - (float)ah1);
      }
#pragma unroll
      for (int ct = 0; ct < 3; ++ct) {
        const int base = ((ct * 2 + h) * 32 + lane) * 8;
        const v16bf bh = __builtin_bit_cast(v16bf, *(const v8u*)&wbuf[cur][0][base]);
        const v16bf bl = __builtin_bit_cast(v16bf, *(const v8u*)&wbuf[cur][1][base]);
        accs[ct] = __builtin_amdgcn_wmma_f32_16x16x32_bf16(false, a_hi, false, bh, (short)0, accs[ct], false, false);
        accs[ct] = __builtin_amdgcn_wmma_f32_16x16x32_bf16(false, a_hi, false, bl, (short)0, accs[ct], false, false);
        accs[ct] = __builtin_amdgcn_wmma_f32_16x16x32_bf16(false, a_lo, false, bh, (short)0, accs[ct], false, false);
      }
    }
  }

#pragma unroll
  for (int ct = 0; ct < 3; ++ct) {
    const int c = ct * 16 + l15;
#pragma unroll
    for (int r = 0; r < 8; ++r) {
      const int tq = t0 + pt * 16 + r + 8 * g;
      if (tq < T_OUT && c < COUT1)
        y1[((size_t)tq * NBATCH + n) * COUT1 + c] = accs[ct][r];
    }
  }
}

// ---------------------------------------------------------------------------
// conv3 GEMM: spikes (bf16-exact A, plain cvt) x pre-packed band matrix B.
// 384 threads = 12 waves = 4 position-tiles x 3 channel-tiles.
// ---------------------------------------------------------------------------
__global__ __launch_bounds__(384) void conv3_wmma(const float* __restrict__ s1,
                                                  const unsigned int* __restrict__ bpk,
                                                  float* __restrict__ y3) {
  const int p0   = blockIdx.x * 64;
  const int tid  = threadIdx.x;
  const int wave = tid >> 5;
  const int lane = tid & 31;
  const int pt   = wave & 3;
  const int ct   = wave >> 2;
  const int g    = lane >> 4;
  const int l15  = lane & 15;

  const int c2 = ct * 16 + l15;
  const int p  = p0 + pt * 16 + l15;
  const float* row = s1 + (size_t)p * COUT1;
  const bool pv = (p < POS);

  v8f acc = {};
#pragma unroll
  for (int h = 0; h < 2; ++h) {
    const int kb = h * 32;
    v16bf a = {};
#pragma unroll
    for (int vv = 0; vv < 8; ++vv) {
      const int ka = kb + ((vv < 4) ? 0 : 16) + g * 8 + 2 * (vv & 3);
      const float a0 = (pv && ka < COUT1) ? row[ka] : 0.0f;
      const float a1 = (pv && ka + 1 < COUT1) ? row[ka + 1] : 0.0f;
      a[2 * vv]     = (__bf16)a0;       // spikes: exact in bf16
      a[2 * vv + 1] = (__bf16)a1;
    }
    const int base = ((ct * 2 + h) * 32 + lane) * 8;
    const v16bf bh = __builtin_bit_cast(v16bf, *(const v8u*)&bpk[base]);
    const v16bf bl = __builtin_bit_cast(v16bf, *(const v8u*)&bpk[W3PK + base]);
    acc = __builtin_amdgcn_wmma_f32_16x16x32_bf16(false, a, false, bh, (short)0, acc, false, false);
    acc = __builtin_amdgcn_wmma_f32_16x16x32_bf16(false, a, false, bl, (short)0, acc, false, false);
  }

#pragma unroll
  for (int r = 0; r < 8; ++r) {
    const int pr = p0 + pt * 16 + r + 8 * g;
    if (pr < POS && c2 < C3)
      y3[(size_t)pr * C3 + c2] = acc[r];
  }
}

// ---------------------------------------------------------------------------
// BN statistics: channel-pinned private partials -> LDS atomics -> global.
// ---------------------------------------------------------------------------
__global__ __launch_bounds__(256) void bn_stats(const float* __restrict__ y, int rows, int C,
                                                float* __restrict__ sum, float* __restrict__ sq) {
  __shared__ float ssum[48], ssq[48];
  const int tid = threadIdx.x;
  if (tid < 48) { ssum[tid] = 0.0f; ssq[tid] = 0.0f; }
  __syncthreads();

  const int slots = 256 / C;
  const int slot  = tid / C;
  const int c     = tid - slot * C;
  if (slot < slots) {
    float a = 0.0f, b = 0.0f;
    const long long stride = (long long)gridDim.x * slots;
    for (long long r = (long long)blockIdx.x * slots + slot; r < rows; r += stride) {
      const float v = y[r * (long long)C + c];
      a += v; b += v * v;
    }
    atomicAdd(&ssum[c], a);
    atomicAdd(&ssq[c], b);
  }
  __syncthreads();
  if (tid < C) {
    atomicAdd(&sum[tid], ssum[tid]);
    atomicAdd(&sq[tid], ssq[tid]);
  }
}

__global__ void bn_params(const float* __restrict__ sum, const float* __restrict__ sq,
                          const float* __restrict__ gamma, const float* __restrict__ beta,
                          float invCount, int C, float* __restrict__ ss) {
  const int c = threadIdx.x;
  if (c < C) {
    const float mean  = sum[c] * invCount;
    const float var   = sq[c] * invCount - mean * mean;
    const float scale = gamma[c] * rsqrtf(var + 1e-5f);
    ss[c]      = scale;
    ss[48 + c] = beta[c] - mean * scale;
  }
}

// PLIF scan: one thread per (n,c) sequence; [t][n][c] layout keeps loads coalesced.
__global__ __launch_bounds__(256) void plif(float* __restrict__ buf, const float* __restrict__ ss,
                                            const float* __restrict__ wparam, int C, int NC, int T) {
  const int idx = blockIdx.x * blockDim.x + threadIdx.x;
  if (idx >= NC) return;
  const int c = idx % C;
  const float scale = ss[c];
  const float shift = ss[48 + c];
  const float decay = 1.0f / (1.0f + expf(-wparam[0]));
  float v = 0.0f;
  for (int t = 0; t < T; ++t) {
    const size_t a = (size_t)t * NC + idx;
    const float yv = buf[a] * scale + shift;
    v += (yv - v) * decay;
    const float s = (v >= 1.0f) ? 1.0f : 0.0f;
    buf[a] = s;
    v = (s != 0.0f) ? 0.0f : v;
  }
}

__global__ __launch_bounds__(64) void mean_fc(const float* __restrict__ s3,
                                              const float* __restrict__ fcw,
                                              const float* __restrict__ fcb,
                                              float* __restrict__ out) {
  __shared__ float feat[C3];
  const int n = blockIdx.x;
  const int c = threadIdx.x;
  if (c < C3) {
    float acc = 0.0f;
    for (int t = 0; t < T_OUT; ++t) acc += s3[((size_t)t * NBATCH + n) * C3 + c];
    feat[c] = acc * (1.0f / (float)T_OUT);
  }
  __syncthreads();
  if (c < 3) {
    float acc = fcb[c];
    for (int k = 0; k < C3; ++k) acc += feat[k] * fcw[c * C3 + k];
    out[n * 3 + c] = acc;
  }
}

__global__ void zero_f32(float* p, int n) {
  const int i = blockIdx.x * blockDim.x + threadIdx.x;
  if (i < n) p[i] = 0.0f;
}

// ---------------------------------------------------------------------------
extern "C" void kernel_launch(void* const* d_in, const int* in_sizes, int n_in,
                              void* d_out, int out_size, void* d_ws, size_t ws_size,
                              hipStream_t stream) {
  (void)in_sizes; (void)n_in; (void)out_size; (void)ws_size;
  const float* x   = (const float*)d_in[0];
  const float* w1c = (const float*)d_in[1];   // conv1_w (40,80,64)
  const float* g1  = (const float*)d_in[3];   // bn1_gamma
  const float* b1  = (const float*)d_in[4];   // bn1_beta
  const float* w1p = (const float*)d_in[5];   // PLIF w1 (scalar)
  const float* w3c = (const float*)d_in[6];   // conv3_w (64 floats)
  const float* g3  = (const float*)d_in[8];   // bn3_gamma
  const float* b3  = (const float*)d_in[9];   // bn3_beta
  const float* w3p = (const float*)d_in[10];  // PLIF w3 (scalar)
  const float* fcw = (const float*)d_in[11];  // fc_w (3,41)
  const float* fcb = (const float*)d_in[12];  // fc_b (3)
  float* out = (float*)d_out;

  float* ws = (float*)d_ws;
  float* y1 = ws;                                   // [469][1024][40] f32
  float* y3 = ws + (size_t)POS * COUT1;             // [469][1024][41] f32
  float* st = y3 + (size_t)POS * C3;                // stats scratch (384 f32)
  float* sum1 = st;        float* sq1 = st + 48;   float* ss1 = st + 96;
  float* sum3 = st + 192;  float* sq3 = st + 240;  float* ss3 = st + 288;
  unsigned int* wpk1 = (unsigned int*)(st + 384);   // 2*W1PK dwords (32B-aligned)
  unsigned int* bpk3 = wpk1 + 2 * W1PK;             // 2*W3PK dwords

  zero_f32<<<1, 384, 0, stream>>>(st, 384);
  pack_w1<<<(W1PK + 255) / 256, 256, 0, stream>>>(w1c, wpk1);
  pack_w3<<<(W3PK + 255) / 256, 256, 0, stream>>>(w3c, bpk3);

  conv1_wmma<<<dim3((T_OUT + 63) / 64, NBATCH), 128, 0, stream>>>(x, wpk1, y1);
  bn_stats<<<512, 256, 0, stream>>>(y1, POS, COUT1, sum1, sq1);
  bn_params<<<1, 64, 0, stream>>>(sum1, sq1, g1, b1, 1.0f / (float)POS, COUT1, ss1);
  plif<<<(NBATCH * COUT1 + 255) / 256, 256, 0, stream>>>(y1, ss1, w1p, COUT1, NBATCH * COUT1, T_OUT);
  conv3_wmma<<<(POS + 63) / 64, 384, 0, stream>>>(y1, bpk3, y3);
  bn_stats<<<512, 256, 0, stream>>>(y3, POS, C3, sum3, sq3);
  bn_params<<<1, 64, 0, stream>>>(sum3, sq3, g3, b3, 1.0f / (float)POS, C3, ss3);
  plif<<<(NBATCH * C3 + 255) / 256, 256, 0, stream>>>(y3, ss3, w3p, C3, NBATCH * C3, T_OUT);
  mean_fc<<<NBATCH, 64, 0, stream>>>(y3, fcw, fcb, out);
}
